// FastVARSelfAttention_3925600109079
// MI455X (gfx1250) — compile-verified, matrix operand order checked by reading, not run on hardware
//
#include <hip/hip_runtime.h>
#include <hip/hip_bf16.h>

// ---------------------------------------------------------------------------
// FastVAR self-attention for gfx1250 (MI455X, wave32, WMMA 16x16x32 f16)
//   B=4, L=1365 (padded 1376), C=1024, H=16, hd=64
// Pipeline:
//   K0: f32 -> f16 pre-convert of x, Wqkv, Wproj (done once, enables pure
//       f16 copies for LDS staging -> GLOBAL_LOAD_ASYNC_TO_LDS path).
//   K1: qkv = x @ Wqkv^T + bias; fused epilogue: cos-norm, per-head temp,
//       RoPE; writes q,k as f16 [B,H,Lp,hd] and v transposed [B,H,hd,Lp].
//       Block tile 128x128, 8 waves, wave tile 32x64 (8 WMMA / k-chunk).
//   K2: flash attention per (b,h,32-row q tile), one wave per tile,
//       online softmax in WMMA C-fragment layout; K/V fragments reused
//       across the two 16-row blocks (16 WMMA / 32-wide kv chunk).
//   K3: out = attn @ Wproj^T + bproj (f32 output), same GEMM shape as K1.
// ---------------------------------------------------------------------------

typedef __attribute__((ext_vector_type(16))) _Float16 v16h;
typedef __attribute__((ext_vector_type(8)))  _Float16 v8h;
typedef __attribute__((ext_vector_type(8)))  float    v8f;
typedef __attribute__((ext_vector_type(4)))  int      v4i;

#define TL    1365          // sequence length
#define TLP   1376          // padded (mult of 32)
#define NH    16
#define HD    64
#define CDIM  1024
#define NB    4
#define MROWS (NB * TL)     // 5460
#define MT    43            // ceil(MROWS/128)
#define QT    43            // ceil(TL/32)
#define KCH   (CDIM / 32)   // 32 K-chunks of 32

#if defined(__has_builtin)
#if __has_builtin(__builtin_amdgcn_global_load_async_to_lds_b128)
#define USE_ASYNC_LDS 1
#endif
#endif
#ifndef USE_ASYNC_LDS
#define USE_ASYNC_LDS 0
#endif

__device__ __forceinline__ v8f wmma16(v16h a, v16h b, v8f c) {
  return __builtin_amdgcn_wmma_f32_16x16x32_f16(
      /*neg_a=*/false, a, /*neg_b=*/false, b,
      /*c_mod=*/(short)0, c, /*reuse_a=*/false, /*reuse_b=*/false);
}

__device__ __forceinline__ v16h cat8(v8h lo, v8h hi) {
  v16h r;
#pragma unroll
  for (int j = 0; j < 8; ++j) { r[j] = lo[j]; r[j + 8] = hi[j]; }
  return r;
}

// Fragment load: K(j) = (j&7) + (j>>3)*16 + khalf*8  (caller folds khalf*8)
__device__ __forceinline__ v16h gfrag(const _Float16* p) {
  v8h lo = *(const v8h*)(p);
  v8h hi = *(const v8h*)(p + 16);
  return cat8(lo, hi);
}

#if USE_ASYNC_LDS
// 16-byte async copy global->LDS (ASYNCcnt-tracked, bypasses VGPRs).
// Builtin takes generic v4i* pointers (per clang-22 diagnostic) + imm offset/cpol.
__device__ __forceinline__ void async_cp16(const _Float16* g, _Float16* l) {
  __builtin_amdgcn_global_load_async_to_lds_b128((v4i*)g, (v4i*)l, 0, 0);
}
__device__ __forceinline__ void async_wait0() {
#if __has_builtin(__builtin_amdgcn_s_wait_asynccnt)
  __builtin_amdgcn_s_wait_asynccnt(0);
#else
  asm volatile("s_wait_asynccnt 0x0" ::: "memory");
#endif
}
#endif

__device__ __forceinline__ float redmax16(float v) {
#pragma unroll
  for (int m = 1; m < 16; m <<= 1) v = fmaxf(v, __shfl_xor(v, m, 32));
  return v;
}
__device__ __forceinline__ float redsum16(float v) {
#pragma unroll
  for (int m = 1; m < 16; m <<= 1) v += __shfl_xor(v, m, 32);
  return v;
}

// block-causal visibility: cum blocks {1,5,21,85,341,1365}
__device__ __forceinline__ int visible_len(int pos) {
  if (pos < 1)   return 1;
  if (pos < 5)   return 5;
  if (pos < 21)  return 21;
  if (pos < 85)  return 85;
  if (pos < 341) return 341;
  return 1365;
}

// ---------------------------------------------------------------------------
// Kernel 0: f32 -> f16 convert, 8 elements per thread
// ---------------------------------------------------------------------------
__global__ __launch_bounds__(256)
void cvt16_kernel(const float* __restrict__ s, _Float16* __restrict__ d, int n8) {
  int i = blockIdx.x * 256 + threadIdx.x;
  if (i < n8) {
    const float4* sp = (const float4*)s + (size_t)i * 2;
    float4 a = sp[0], b = sp[1];
    v8h h;
    h[0]=(_Float16)a.x; h[1]=(_Float16)a.y; h[2]=(_Float16)a.z; h[3]=(_Float16)a.w;
    h[4]=(_Float16)b.x; h[5]=(_Float16)b.y; h[6]=(_Float16)b.z; h[7]=(_Float16)b.w;
    *(v8h*)(d + (size_t)i * 8) = h;
  }
}

// ---------------------------------------------------------------------------
// Shared GEMM staging: 128x32 f16 tile, 256 threads, 2 x b128 per thread
// (async path goes straight to LDS; fallback is reg copy)
// ---------------------------------------------------------------------------
__device__ __forceinline__ void stage_tile(const _Float16* __restrict__ gsrc,
                                           _Float16* __restrict__ ldst) {
#if USE_ASYNC_LDS
  async_cp16(gsrc, ldst);
  async_cp16(gsrc + 8, ldst + 8);
#else
  *(v8h*)ldst       = *(const v8h*)gsrc;
  *(v8h*)(ldst + 8) = *(const v8h*)(gsrc + 8);
#endif
}

// ---------------------------------------------------------------------------
// Kernel 1: QKV GEMM + fused norm/RoPE epilogue
// grid (43, 24), block 256 (8 wave32); block tile 128x128, wave tile 32x64
// ---------------------------------------------------------------------------
__global__ __launch_bounds__(256)
void qkv_gemm_kernel(const _Float16* __restrict__ x16, const _Float16* __restrict__ w16,
                     const float* __restrict__ qbias, const float* __restrict__ vbias,
                     const float* __restrict__ smlog,
                     const float* __restrict__ rc, const float* __restrict__ rs,
                     _Float16* __restrict__ qo, _Float16* __restrict__ ko,
                     _Float16* __restrict__ vto) {
  __shared__ __align__(16) _Float16 As[128][40];
  __shared__ __align__(16) _Float16 Bs[128][40];

  const int tid    = threadIdx.x;
  const int lane   = tid & 31;
  const int wave   = tid >> 5;
  const int lane15 = lane & 15;
  const int khalf  = lane >> 4;
  const int wm     = wave & 3;   // 4 M-strips of 32
  const int wn     = wave >> 2;  // 2 N-strips of 64
  const int mb     = blockIdx.x; // 43
  const int nb     = blockIdx.y; // 24

  // staging roles: 128 rows, 2 threads/row, 16 halfs each
  const int srow = tid >> 1, skq = (tid & 1) * 16;
  int ag = mb * 128 + srow;
  if (ag >= MROWS) ag = MROWS - 1;  // clamp: rows beyond M are never stored
  const int bn = nb * 128 + srow;

  const _Float16* gA = x16 + (size_t)ag * CDIM + skq;
  const _Float16* gB = w16 + (size_t)bn * CDIM + skq;
  _Float16* lA = &As[srow][skq];
  _Float16* lB = &Bs[srow][skq];

  v8f acc[2][4] = {{{}, {}, {}, {}}, {{}, {}, {}, {}}};

  for (int kc = 0; kc < KCH; ++kc) {
    __syncthreads();
    stage_tile(gA + kc * 32, lA);
    stage_tile(gB + kc * 32, lB);
    if (kc + 1 < KCH) {
      __builtin_prefetch(gA + kc * 32 + 32, 0, 0);
      __builtin_prefetch(gB + kc * 32 + 32, 0, 0);
    }
#if USE_ASYNC_LDS
    async_wait0();
#endif
    __syncthreads();

    // issue all fragment loads, then burst 8 WMMAs
    v16h af0 = gfrag(&As[wm * 32 + lane15][khalf * 8]);
    v16h af1 = gfrag(&As[wm * 32 + 16 + lane15][khalf * 8]);
    v16h bf0 = gfrag(&Bs[wn * 64 + lane15][khalf * 8]);
    v16h bf1 = gfrag(&Bs[wn * 64 + 16 + lane15][khalf * 8]);
    v16h bf2 = gfrag(&Bs[wn * 64 + 32 + lane15][khalf * 8]);
    v16h bf3 = gfrag(&Bs[wn * 64 + 48 + lane15][khalf * 8]);
    acc[0][0] = wmma16(af0, bf0, acc[0][0]);
    acc[0][1] = wmma16(af0, bf1, acc[0][1]);
    acc[0][2] = wmma16(af0, bf2, acc[0][2]);
    acc[0][3] = wmma16(af0, bf3, acc[0][3]);
    acc[1][0] = wmma16(af1, bf0, acc[1][0]);
    acc[1][1] = wmma16(af1, bf1, acc[1][1]);
    acc[1][2] = wmma16(af1, bf2, acc[1][2]);
    acc[1][3] = wmma16(af1, bf3, acc[1][3]);
  }

  // ---- fused epilogue (wave strip = 32 rows x one head's 64 cols) ----
  const int colb = nb * 128 + wn * 64;   // multiple of 64 -> one head section
  const int sec  = colb >> 10;           // 0=q, 1=k, 2=v
  const int head = (colb & 1023) >> 6;
  const int roff = khalf * 8;

  float vals[2][4][8];
#pragma unroll
  for (int r = 0; r < 2; ++r)
#pragma unroll
    for (int t = 0; t < 4; ++t)
#pragma unroll
      for (int i = 0; i < 8; ++i) vals[r][t][i] = acc[r][t][i];

  int  bb[2][8], pos[2][8];
  bool valid[2][8];
#pragma unroll
  for (int r = 0; r < 2; ++r)
#pragma unroll
    for (int i = 0; i < 8; ++i) {
      int g = mb * 128 + wm * 32 + r * 16 + roff + i;
      valid[r][i] = (g < MROWS);
      int gb = g / TL;  if (gb > NB - 1) gb = NB - 1;
      bb[r][i]  = gb;
      pos[r][i] = g - gb * TL;  if (pos[r][i] > TL - 1) pos[r][i] = TL - 1;
    }

  if (sec != 1) {  // bias add for q and v sections
    const float* bsrc = (sec == 0) ? qbias : vbias;
#pragma unroll
    for (int t = 0; t < 4; ++t) {
      float bv = bsrc[head * HD + t * 16 + lane15];
#pragma unroll
      for (int r = 0; r < 2; ++r)
#pragma unroll
        for (int i = 0; i < 8; ++i) vals[r][t][i] += bv;
    }
  }

  if (sec < 2) {
    float mulh = 1.0f;
    if (sec == 0) mulh = __expf(fminf(smlog[head], 4.6051702f)); // log(100)
#pragma unroll
    for (int r = 0; r < 2; ++r)
#pragma unroll
      for (int i = 0; i < 8; ++i) {
        float ss = 0.f;
#pragma unroll
        for (int t = 0; t < 4; ++t) ss += vals[r][t][i] * vals[r][t][i];
        ss = redsum16(ss);
        float inv = mulh / fmaxf(sqrtf(ss), 1e-12f);
#pragma unroll
        for (int t = 0; t < 4; ++t) vals[r][t][i] *= inv;
      }
    // RoPE: rotate consecutive pairs; partner is adjacent lane
#pragma unroll
    for (int t = 0; t < 4; ++t) {
      int j = (t * 16 + lane15) >> 1;
#pragma unroll
      for (int r = 0; r < 2; ++r)
#pragma unroll
        for (int i = 0; i < 8; ++i) {
          float cw = rc[pos[r][i] * 32 + j];
          float sw = rs[pos[r][i] * 32 + j];
          float other = __shfl_xor(vals[r][t][i], 1, 32);
          float v = vals[r][t][i];
          vals[r][t][i] = (lane & 1) ? (sw * other + cw * v)
                                     : (cw * v - sw * other);
        }
    }
    _Float16* dst = (sec == 0) ? qo : ko;   // [B,H,Lp,hd]
#pragma unroll
    for (int t = 0; t < 4; ++t) {
      int c = t * 16 + lane15;
#pragma unroll
      for (int r = 0; r < 2; ++r)
#pragma unroll
        for (int i = 0; i < 8; ++i)
          if (valid[r][i]) {
            size_t idx = (((size_t)bb[r][i] * NH + head) * TLP + pos[r][i]) * HD + c;
            dst[idx] = (_Float16)vals[r][t][i];
          }
    }
  } else {  // v transposed: [B,H,hd,Lp]
#pragma unroll
    for (int t = 0; t < 4; ++t) {
      int c = t * 16 + lane15;
#pragma unroll
      for (int r = 0; r < 2; ++r)
#pragma unroll
        for (int i = 0; i < 8; ++i)
          if (valid[r][i]) {
            size_t idx = (((size_t)bb[r][i] * NH + head) * HD + c) * TLP + pos[r][i];
            vto[idx] = (_Float16)vals[r][t][i];
          }
    }
  }
}

// ---------------------------------------------------------------------------
// Kernel 2: flash attention, one wave per (b, h, 32-query-row tile)
// grid 43*16*4, block 32
// ---------------------------------------------------------------------------
__global__ __launch_bounds__(32)
void attn_kernel(const _Float16* __restrict__ q, const _Float16* __restrict__ k,
                 const _Float16* __restrict__ vt, _Float16* __restrict__ ao) {
  __shared__ __align__(16) _Float16 Ps[32][40];

  const int lane   = threadIdx.x & 31;
  const int lane15 = lane & 15;
  const int khalf  = lane >> 4;
  const int roff   = khalf * 8;

  int bid = blockIdx.x;
  const int qt = bid % QT;  bid /= QT;
  const int h  = bid % NH;
  const int b  = bid / NH;
  const int qbase = qt * 32;
  const size_t bh = (size_t)b * NH + h;

  // persistent Q fragments: 2 row-blocks x 2 hd-chunks
  const _Float16* qp0 = q + (bh * TLP + (qbase + lane15)) * HD;
  const _Float16* qp1 = q + (bh * TLP + (qbase + 16 + lane15)) * HD;
  v16h qf00 = gfrag(qp0 + khalf * 8);
  v16h qf01 = gfrag(qp0 + 32 + khalf * 8);
  v16h qf10 = gfrag(qp1 + khalf * 8);
  v16h qf11 = gfrag(qp1 + 32 + khalf * 8);

  int vis[2][8];
#pragma unroll
  for (int r = 0; r < 2; ++r)
#pragma unroll
    for (int i = 0; i < 8; ++i) {
      int p = qbase + r * 16 + roff + i;
      vis[r][i] = visible_len(p < TL ? p : TL - 1);
    }

  float m8[2][8], l8[2][8];
  v8f O[2][4] = {{{}, {}, {}, {}}, {{}, {}, {}, {}}};
#pragma unroll
  for (int r = 0; r < 2; ++r)
#pragma unroll
    for (int i = 0; i < 8; ++i) { m8[r][i] = -3e38f; l8[r][i] = 0.f; }

  const int vmax = visible_len((qbase + 31 < TL) ? (qbase + 31) : (TL - 1));
  const int nch  = (vmax + 31) >> 5;

  const _Float16* kbase = k + bh * TLP * HD;
  const _Float16* vbase = vt + bh * HD * TLP;

  for (int cc = 0; cc < nch; ++cc) {
    const int kvb = cc * 32;

    // scores: issue all K fragment loads, then 8 WMMAs (K frags reused x2)
    const _Float16* kp0 = kbase + (size_t)(kvb + lane15) * HD;
    const _Float16* kp1 = kbase + (size_t)(kvb + 16 + lane15) * HD;
    v16h kA0 = gfrag(kp0 + khalf * 8);
    v16h kB0 = gfrag(kp0 + 32 + khalf * 8);
    v16h kA1 = gfrag(kp1 + khalf * 8);
    v16h kB1 = gfrag(kp1 + 32 + khalf * 8);
    v8f S[2][2] = {{{}, {}}, {{}, {}}};
    S[0][0] = wmma16(qf00, kA0, S[0][0]);
    S[0][0] = wmma16(qf01, kB0, S[0][0]);
    S[0][1] = wmma16(qf00, kA1, S[0][1]);
    S[0][1] = wmma16(qf01, kB1, S[0][1]);
    S[1][0] = wmma16(qf10, kA0, S[1][0]);
    S[1][0] = wmma16(qf11, kB0, S[1][0]);
    S[1][1] = wmma16(qf10, kA1, S[1][1]);
    S[1][1] = wmma16(qf11, kB1, S[1][1]);

    // V fragments (shared by both row blocks) -- overlap with softmax VALU
    v16h vf[4];
#pragma unroll
    for (int t = 0; t < 4; ++t)
      vf[t] = gfrag(vbase + (size_t)(t * 16 + lane15) * TLP + kvb + khalf * 8);

    // online softmax in C-fragment layout (row stats are lane-replicated)
#pragma unroll
    for (int r = 0; r < 2; ++r)
#pragma unroll
      for (int i = 0; i < 8; ++i) {
        bool msk0 = (kvb + lane15)      >= vis[r][i];
        bool msk1 = (kvb + 16 + lane15) >= vis[r][i];
        float a0 = msk0 ? -3e38f : S[r][0][i];
        float a1 = msk1 ? -3e38f : S[r][1][i];
        float rm = redmax16(fmaxf(a0, a1));
        float mn = fmaxf(m8[r][i], rm);
        float sc = __expf(m8[r][i] - mn);
        float p0 = msk0 ? 0.f : __expf(S[r][0][i] - mn);
        float p1 = msk1 ? 0.f : __expf(S[r][1][i] - mn);
        l8[r][i] = l8[r][i] * sc + redsum16(p0 + p1);
        m8[r][i] = mn;
        O[r][0][i] *= sc; O[r][1][i] *= sc; O[r][2][i] *= sc; O[r][3][i] *= sc;
        Ps[r * 16 + roff + i][lane15]      = (_Float16)p0;
        Ps[r * 16 + roff + i][16 + lane15] = (_Float16)p1;
      }
    __syncthreads();  // single-wave WG: cheap; orders LDS write -> read

    v16h pf0 = gfrag(&Ps[lane15][khalf * 8]);
    v16h pf1 = gfrag(&Ps[16 + lane15][khalf * 8]);
#pragma unroll
    for (int t = 0; t < 4; ++t) {
      O[0][t] = wmma16(pf0, vf[t], O[0][t]);
      O[1][t] = wmma16(pf1, vf[t], O[1][t]);
    }
    __syncthreads();
  }

  // normalize + store f16 [B, L, C] with C = H*hd
#pragma unroll
  for (int r = 0; r < 2; ++r)
#pragma unroll
    for (int i = 0; i < 8; ++i) {
      int p = qbase + r * 16 + roff + i;
      float inv = 1.0f / l8[r][i];
      if (p < TL) {
        _Float16* op = ao + ((size_t)b * TL + p) * CDIM + h * HD;
#pragma unroll
        for (int t = 0; t < 4; ++t)
          op[t * 16 + lane15] = (_Float16)(O[r][t][i] * inv);
      }
    }
}

// ---------------------------------------------------------------------------
// Kernel 3: output projection, f16 A x f16 W, f32 out + bias
// grid (43, 8), block 256; same tiling as K1
// ---------------------------------------------------------------------------
__global__ __launch_bounds__(256)
void proj_gemm_kernel(const _Float16* __restrict__ a, const _Float16* __restrict__ w16,
                      const float* __restrict__ bias, float* __restrict__ out) {
  __shared__ __align__(16) _Float16 As[128][40];
  __shared__ __align__(16) _Float16 Bs[128][40];

  const int tid    = threadIdx.x;
  const int lane   = tid & 31;
  const int wave   = tid >> 5;
  const int lane15 = lane & 15;
  const int khalf  = lane >> 4;
  const int wm     = wave & 3;
  const int wn     = wave >> 2;
  const int mb     = blockIdx.x;  // 43
  const int nb     = blockIdx.y;  // 8

  const int srow = tid >> 1, skq = (tid & 1) * 16;
  int ag = mb * 128 + srow;
  if (ag >= MROWS) ag = MROWS - 1;
  const int bn = nb * 128 + srow;

  const _Float16* gA = a   + (size_t)ag * CDIM + skq;
  const _Float16* gB = w16 + (size_t)bn * CDIM + skq;
  _Float16* lA = &As[srow][skq];
  _Float16* lB = &Bs[srow][skq];

  v8f acc[2][4] = {{{}, {}, {}, {}}, {{}, {}, {}, {}}};

  for (int kc = 0; kc < KCH; ++kc) {
    __syncthreads();
    stage_tile(gA + kc * 32, lA);
    stage_tile(gB + kc * 32, lB);
    if (kc + 1 < KCH) {
      __builtin_prefetch(gA + kc * 32 + 32, 0, 0);
      __builtin_prefetch(gB + kc * 32 + 32, 0, 0);
    }
#if USE_ASYNC_LDS
    async_wait0();
#endif
    __syncthreads();

    v16h af0 = gfrag(&As[wm * 32 + lane15][khalf * 8]);
    v16h af1 = gfrag(&As[wm * 32 + 16 + lane15][khalf * 8]);
    v16h bf0 = gfrag(&Bs[wn * 64 + lane15][khalf * 8]);
    v16h bf1 = gfrag(&Bs[wn * 64 + 16 + lane15][khalf * 8]);
    v16h bf2 = gfrag(&Bs[wn * 64 + 32 + lane15][khalf * 8]);
    v16h bf3 = gfrag(&Bs[wn * 64 + 48 + lane15][khalf * 8]);
    acc[0][0] = wmma16(af0, bf0, acc[0][0]);
    acc[0][1] = wmma16(af0, bf1, acc[0][1]);
    acc[0][2] = wmma16(af0, bf2, acc[0][2]);
    acc[0][3] = wmma16(af0, bf3, acc[0][3]);
    acc[1][0] = wmma16(af1, bf0, acc[1][0]);
    acc[1][1] = wmma16(af1, bf1, acc[1][1]);
    acc[1][2] = wmma16(af1, bf2, acc[1][2]);
    acc[1][3] = wmma16(af1, bf3, acc[1][3]);
  }

  const int roff = khalf * 8;
#pragma unroll
  for (int t = 0; t < 4; ++t) {
    int col = nb * 128 + wn * 64 + t * 16 + lane15;
    float bv = bias[col];
#pragma unroll
    for (int r = 0; r < 2; ++r)
#pragma unroll
      for (int i = 0; i < 8; ++i) {
        int g = mb * 128 + wm * 32 + r * 16 + roff + i;
        if (g < MROWS) out[(size_t)g * CDIM + col] = acc[r][t][i] + bv;
      }
  }
}

// ---------------------------------------------------------------------------
extern "C" void kernel_launch(void* const* d_in, const int* in_sizes, int n_in,
                              void* d_out, int out_size, void* d_ws, size_t ws_size,
                              hipStream_t stream) {
  const float* x      = (const float*)d_in[0];
  // d_in[1] = attn_bias (recomputed analytically on device)
  const float* rc     = (const float*)d_in[2];
  const float* rs     = (const float*)d_in[3];
  const float* Wqkv   = (const float*)d_in[4];
  const float* qbias  = (const float*)d_in[5];
  const float* vbias  = (const float*)d_in[6];
  const float* smlog  = (const float*)d_in[7];
  const float* Wproj  = (const float*)d_in[8];
  const float* bproj  = (const float*)d_in[9];
  float* out          = (float*)d_out;

  char* w = (char*)d_ws;
  const size_t QKB  = (size_t)NB * NH * TLP * HD * sizeof(_Float16);  // q/k/vT
  const size_t XEL  = (size_t)NB * TL * CDIM;                          // x / attn elems
  const size_t WQEL = (size_t)3 * CDIM * CDIM;
  const size_t WPEL = (size_t)CDIM * CDIM;

  _Float16* qf   = (_Float16*)(w);
  _Float16* kf   = (_Float16*)(w + QKB);
  _Float16* vtf  = (_Float16*)(w + 2 * QKB);
  _Float16* af   = (_Float16*)(w + 3 * QKB);
  _Float16* x16  = af  + XEL;
  _Float16* wq16 = x16 + XEL;
  _Float16* wp16 = wq16 + WQEL;

  // K0: one-time f32 -> f16 conversions (enables async f16 LDS staging)
  cvt16_kernel<<<(int)((XEL / 8 + 255) / 256), 256, 0, stream>>>(x, x16, (int)(XEL / 8));
  cvt16_kernel<<<(int)((WQEL / 8 + 255) / 256), 256, 0, stream>>>(Wqkv, wq16, (int)(WQEL / 8));
  cvt16_kernel<<<(int)((WPEL / 8 + 255) / 256), 256, 0, stream>>>(Wproj, wp16, (int)(WPEL / 8));

  qkv_gemm_kernel<<<dim3(MT, 24), 256, 0, stream>>>(
      x16, wq16, qbias, vbias, smlog, rc, rs, qf, kf, vtf);

  attn_kernel<<<QT * NH * NB, 32, 0, stream>>>(qf, kf, vtf, af);

  proj_gemm_kernel<<<dim3(MT, 8), 256, 0, stream>>>(af, wp16, bproj, out);
}
